// SubgraphAttentionGNN_43078521979063
// MI455X (gfx1250) — matrix operand order repaired and checked by reading, not compiled, not run on hardware
//
#include <hip/hip_runtime.h>
#include <cstdint>
#include <cstddef>

// ---------------------------------------------------------------------------
// Problem constants (match reference)
// ---------------------------------------------------------------------------
#define NNODES 20000
#define NEDGES 320000
#define ETOT   (NEDGES + NNODES)   // self-loops appended
#define INCH   128
#define HEADS  4
#define CH     64
#define HID    256                 // HEADS*CH
#define BATCH  4096

typedef __attribute__((ext_vector_type(16))) __bf16        v16bf;
typedef __attribute__((ext_vector_type(8)))  float         v8f;
typedef __attribute__((ext_vector_type(4)))  unsigned int  v4u;

static __device__ __forceinline__ float lrelu02(float v) { return v > 0.f ? v : 0.2f * v; }

// ---------------------------------------------------------------------------
// Utility kernels
// ---------------------------------------------------------------------------
__global__ void fill_f32(float* __restrict__ p, float v, int n) {
  int i = blockIdx.x * blockDim.x + threadIdx.x;
  if (i < n) p[i] = v;
}

__global__ void cvt_bf16(const float* __restrict__ in, __bf16* __restrict__ out, int n) {
  int i = blockIdx.x * blockDim.x + threadIdx.x;
  if (i < n) out[i] = (__bf16)in[i];
}

// Convert f32 weight matrix B[K x Nc] (row-major) into WMMA-fragment-packed
// bf16: Bp[(k/32)*Nc + n][32] where the 32-slice is k%32 in natural order.
// A lane (column n, half h) then reads its whole 16-value B fragment for one
// 16x16x32 WMMA step as 16 consecutive bf16 = two b128 loads.
__global__ void cvt_bpack(const float* __restrict__ Bsrc, __bf16* __restrict__ Bp,
                          int Kk, int Nc) {
  int id = blockIdx.x * blockDim.x + threadIdx.x;
  if (id >= Kk * Nc) return;
  const int k = id / Nc, n = id % Nc;
  const int t = k >> 5, kr = k & 31;
  Bp[(((size_t)t * Nc + n) << 5) + kr] = (__bf16)Bsrc[id];
}

// ---------------------------------------------------------------------------
// WMMA GEMM:  Cf[M x NC] (f32) = A[M x K](bf16 row-major) * B[K x NC](packed bf16)
// K and NC are compile-time -> fully unrolled v_wmma_f32_16x16x32_bf16 chain and
// all A/B/C addresses become immediate offsets from per-lane base pointers
// (no per-iteration 64-bit VALU address math, minimal WMMA->VALU hazards).
// Each wave computes a 32x16 output tile (2 M-tiles sharing one B fragment);
// 4 waves/block cover a 32x64 strip. M%32==0, NC%64==0, K%32==0 all hold here,
// so EXEC is all-ones everywhere (WMMA requirement).
// A fragment = 2 x global_load_b128, B fragment = 2 x global_load_b128 per step.
// ---------------------------------------------------------------------------
template <int K, int NC>
__global__ void __launch_bounds__(128)
wmma_gemm_bf16(const __bf16* __restrict__ A, const __bf16* __restrict__ Bp,
               float* __restrict__ Cf, __bf16* __restrict__ Cb,
               const float* __restrict__ bias, int act)
{
  const int lane = threadIdx.x & 31;
  const int wave = threadIdx.x >> 5;
  const int l16  = lane & 15;
  const int hi   = lane >> 4;            // 0: lanes 0-15, 1: lanes 16-31

  const int row0 = blockIdx.y << 5;                    // 32 rows per block
  const int col0 = (blockIdx.x << 6) + (wave << 4);    // 16 cols per wave
  const int n    = col0 + l16;

  // A fragment addressing (16-bit A 16x32): lane half 'hi' holds
  // k in [kbA, kbA+15], contiguous in memory (matches packed B).
  const int kbA = hi << 4;               // 0 or 16
  const __bf16* Arow0 = A + (size_t)(row0 + l16) * K + kbA;
  const __bf16* Arow1 = Arow0 + (size_t)16 * K;

  // Packed-B fragment: 16 consecutive bf16 at ((t*NC + n)*32 + hi*16); per-lane
  // base (n, hi) is fixed, the per-step displacement t*NC*32 is a constant.
  const __bf16* Brow = Bp + (((size_t)n) << 5) + (hi << 4);

  v8f c0 = {};
  v8f c1 = {};

#pragma unroll
  for (int t = 0; t < K / 32; ++t) {
    const int kk = t * 32;
    __builtin_prefetch(Arow0 + kk + 32, 0, 3);   // next K-tile into near caches
    __builtin_prefetch(Arow1 + kk + 32, 0, 3);

    union { v16bf v; v4u q[2]; } bfr;
    bfr.q[0] = *reinterpret_cast<const v4u*>(Brow + (size_t)t * NC * 32);
    bfr.q[1] = *reinterpret_cast<const v4u*>(Brow + (size_t)t * NC * 32 + 8);

    union { v16bf v; v4u q[2]; } a0;
    a0.q[0] = *reinterpret_cast<const v4u*>(Arow0 + kk);
    a0.q[1] = *reinterpret_cast<const v4u*>(Arow0 + kk + 8);
    c0 = __builtin_amdgcn_wmma_f32_16x16x32_bf16(false, a0.v, false, bfr.v,
                                                 (short)0, c0, false, false);

    union { v16bf v; v4u q[2]; } a1;
    a1.q[0] = *reinterpret_cast<const v4u*>(Arow1 + kk);
    a1.q[1] = *reinterpret_cast<const v4u*>(Arow1 + kk + 8);
    c1 = __builtin_amdgcn_wmma_f32_16x16x32_bf16(false, a1.v, false, bfr.v,
                                                 (short)0, c1, false, false);
  }

  const float badd = bias ? bias[n] : 0.0f;
  // C/D layout: VGPR r -> row r (lanes 0-15) / row r+8 (lanes 16-31)
  float*  cf = Cf + (size_t)(row0 + (hi << 3)) * NC + n;
  __bf16* cb = Cb ? (Cb + (size_t)(row0 + (hi << 3)) * NC + n) : nullptr;
#pragma unroll
  for (int mt = 0; mt < 2; ++mt) {
    const v8f& c = mt ? c1 : c0;
#pragma unroll
    for (int r = 0; r < 8; ++r) {
      const size_t o = (size_t)(mt * 16 + r) * NC;
      float v = c[r] + badd;
      if (act) v = fmaxf(v, 0.0f);
      cf[o] = v;
      if (cb) cb[o] = (__bf16)v;
    }
  }
}

// ---------------------------------------------------------------------------
// GAT attention logits: es[n,h] = sum_c h[n,h,c]*a_src[h,c]; ed likewise.
// ---------------------------------------------------------------------------
__global__ void gat_logits(const float* __restrict__ h, const float* __restrict__ as,
                           const float* __restrict__ ad,
                           float* __restrict__ es, float* __restrict__ ed)
{
  int id = blockIdx.x * blockDim.x + threadIdx.x;
  if (id >= NNODES * HEADS) return;
  const int nidx = id / HEADS, hh = id % HEADS;
  const float* hp = h + (size_t)nidx * HID + hh * CH;
  const float* ap = as + hh * CH;
  const float* dp = ad + hh * CH;
  float s = 0.f, d = 0.f;
#pragma unroll 8
  for (int c = 0; c < CH; ++c) { float v = hp[c]; s += v * ap[c]; d += v * dp[c]; }
  es[id] = s;
  ed[id] = d;
}

static __device__ __forceinline__ void edge_sd(const int* __restrict__ ei, int e, int& s, int& d) {
  if (e < NEDGES) { s = ei[e]; d = ei[NEDGES + e]; }
  else            { s = d = e - NEDGES; }            // appended self-loop
}

static __device__ __forceinline__ void atomicMaxF(float* addr, float val) {
  unsigned int* ua  = (unsigned int*)addr;
  unsigned int  old = *ua;
  while (__uint_as_float(old) < val) {
    unsigned int assumed = old;
    old = atomicCAS(ua, assumed, __float_as_uint(val));
    if (old == assumed) break;
  }
}

// Pass 1: segment max of leaky-relu logits per destination/head
__global__ void edge_max(const int* __restrict__ ei, const float* __restrict__ es,
                         const float* __restrict__ ed, float* __restrict__ m)
{
  int id = blockIdx.x * blockDim.x + threadIdx.x;
  if (id >= ETOT * HEADS) return;
  const int e = id / HEADS, hh = id % HEADS;
  int s, d; edge_sd(ei, e, s, d);
  float v = lrelu02(es[s * HEADS + hh] + ed[d * HEADS + hh]);
  atomicMaxF(&m[d * HEADS + hh], v);
}

// Pass 2: segment sum of exp(e - max)
__global__ void edge_den(const int* __restrict__ ei, const float* __restrict__ es,
                         const float* __restrict__ ed, const float* __restrict__ m,
                         float* __restrict__ den)
{
  int id = blockIdx.x * blockDim.x + threadIdx.x;
  if (id >= ETOT * HEADS) return;
  const int e = id / HEADS, hh = id % HEADS;
  int s, d; edge_sd(ei, e, s, d);
  float v = lrelu02(es[s * HEADS + hh] + ed[d * HEADS + hh]);
  atomicAdd(&den[d * HEADS + hh], __expf(v - m[d * HEADS + hh]));
}

// Pass 3: agg[dst,f] += msg[src,f] * alpha(e, head(f))
__global__ void edge_agg(const int* __restrict__ ei, const float* __restrict__ es,
                         const float* __restrict__ ed, const float* __restrict__ m,
                         const float* __restrict__ den, const float* __restrict__ msg,
                         float* __restrict__ agg)
{
  int id = blockIdx.x * blockDim.x + threadIdx.x;
  if (id >= ETOT * HID) return;
  const int e = id / HID, f = id % HID, hh = f / CH;
  int s, d; edge_sd(ei, e, s, d);
  const int ih = d * HEADS + hh;
  float v = lrelu02(es[s * HEADS + hh] + ed[ih]);
  float w = __expf(v - m[ih]);
  float alpha = w / (den[ih] + 1e-16f);
  atomicAdd(&agg[(size_t)d * HID + f], msg[(size_t)s * HID + f] * alpha);
}

// In-place bias (+ optional ReLU) on agg; also emit bf16 copy for the next GEMM.
__global__ void bias_act_cvt(float* __restrict__ agg, const float* __restrict__ bias,
                             __bf16* __restrict__ outb, int act)
{
  int id = blockIdx.x * blockDim.x + threadIdx.x;
  if (id >= NNODES * HID) return;
  float v = agg[id] + bias[id % HID];
  if (act) v = fmaxf(v, 0.f);
  agg[id]  = v;
  outb[id] = (__bf16)v;
}

// z[b] = concat(h[u[b]], h[v[b]]) in bf16
__global__ void gather_pairs(const float* __restrict__ h, const int* __restrict__ u,
                             const int* __restrict__ v, __bf16* __restrict__ zb)
{
  int id = blockIdx.x * blockDim.x + threadIdx.x;
  if (id >= BATCH * 2 * HID) return;
  const int b = id / (2 * HID), f = id % (2 * HID);
  const int node = (f < HID) ? u[b] : v[b];
  const int feat = f & (HID - 1);
  float val = (node >= 0 && node < NNODES) ? h[(size_t)node * HID + feat] : 0.f;
  zb[id] = (__bf16)val;
}

// Final 128 -> 1 head: dot + bias + sigmoid + mask
__global__ void edge_head(const float* __restrict__ z2, const float* __restrict__ w,
                          const float* __restrict__ b, const int* __restrict__ u,
                          const int* __restrict__ v, float* __restrict__ out)
{
  int bi = blockIdx.x * blockDim.x + threadIdx.x;
  if (bi >= BATCH) return;
  const float* zp = z2 + (size_t)bi * (HID / 2);
  float acc = b[0];
#pragma unroll 8
  for (int i = 0; i < HID / 2; ++i) acc += zp[i] * w[i];
  float p = 1.0f / (1.0f + __expf(-acc));
  out[bi] = (u[bi] >= 0 && v[bi] >= 0) ? p : 0.0f;
}

// ---------------------------------------------------------------------------
// Host-side launch
// ---------------------------------------------------------------------------
extern "C" void kernel_launch(void* const* d_in, const int* in_sizes, int n_in,
                              void* d_out, int out_size, void* d_ws, size_t ws_size,
                              hipStream_t stream)
{
  (void)in_sizes; (void)n_in; (void)out_size; (void)ws_size;

  const float* x  = (const float*)d_in[0];
  const int*   ei = (const int*)d_in[1];
  const int*   u  = (const int*)d_in[2];
  const int*   v  = (const int*)d_in[3];
  const float* W[3]  = {(const float*)d_in[4],  (const float*)d_in[8],  (const float*)d_in[12]};
  const float* As[3] = {(const float*)d_in[5],  (const float*)d_in[9],  (const float*)d_in[13]};
  const float* Ad[3] = {(const float*)d_in[6],  (const float*)d_in[10], (const float*)d_in[14]};
  const float* Bi[3] = {(const float*)d_in[7],  (const float*)d_in[11], (const float*)d_in[15]};
  const float* ew0 = (const float*)d_in[16]; const float* eb0 = (const float*)d_in[17];
  const float* ew1 = (const float*)d_in[18]; const float* eb1 = (const float*)d_in[19];
  const float* ew2 = (const float*)d_in[20]; const float* eb2 = (const float*)d_in[21];
  float* out = (float*)d_out;

  // Workspace layout (~66 MB total, 256B-aligned segments)
  char*  base = (char*)d_ws;
  size_t off  = 0;
  auto alloc = [&](size_t bytes) -> void* {
    void* p = base + off;
    off = (off + bytes + 255) & ~(size_t)255;
    return p;
  };
  __bf16* W0b  = (__bf16*)alloc((size_t)INCH * HID * 2);        // packed
  __bf16* W1b  = (__bf16*)alloc((size_t)HID * HID * 2);         // packed
  __bf16* W2b  = (__bf16*)alloc((size_t)HID * HID * 2);         // packed
  __bf16* ew0b = (__bf16*)alloc((size_t)2 * HID * HID * 2);     // packed
  __bf16* ew1b = (__bf16*)alloc((size_t)HID * (HID / 2) * 2);   // packed
  __bf16* xb   = (__bf16*)alloc((size_t)NNODES * HID * 2);      // bf16 layer input (N x K)
  float*  msg  = (float*) alloc((size_t)NNODES * HID * 4);      // h = xW (messages)
  float*  agg  = (float*) alloc((size_t)NNODES * HID * 4);      // aggregated output / final h
  float*  es   = (float*) alloc((size_t)NNODES * HEADS * 4);
  float*  ed   = (float*) alloc((size_t)NNODES * HEADS * 4);
  float*  mbuf = (float*) alloc((size_t)NNODES * HEADS * 4);
  float*  den  = (float*) alloc((size_t)NNODES * HEADS * 4);
  __bf16* zb   = (__bf16*)alloc((size_t)BATCH * 2 * HID * 2);
  float*  z1f  = (float*) alloc((size_t)BATCH * HID * 4);
  __bf16* z1b  = (__bf16*)alloc((size_t)BATCH * HID * 2);
  float*  z2f  = (float*) alloc((size_t)BATCH * (HID / 2) * 4);

  const int TB = 256;
  auto nb = [](long long n, int tb) -> unsigned { return (unsigned)((n + tb - 1) / tb); };

  // Weight conversions into WMMA-fragment-packed bf16; features plain bf16.
  cvt_bpack<<<nb(INCH * HID, TB), TB, 0, stream>>>(W[0], W0b, INCH, HID);
  cvt_bpack<<<nb(HID * HID, TB), TB, 0, stream>>>(W[1], W1b, HID, HID);
  cvt_bpack<<<nb(HID * HID, TB), TB, 0, stream>>>(W[2], W2b, HID, HID);
  cvt_bpack<<<nb(2 * HID * HID, TB), TB, 0, stream>>>(ew0, ew0b, 2 * HID, HID);
  cvt_bpack<<<nb(HID * (HID / 2), TB), TB, 0, stream>>>(ew1, ew1b, HID, HID / 2);
  cvt_bf16<<<nb((long long)NNODES * INCH, TB), TB, 0, stream>>>(x, xb, NNODES * INCH);

  for (int l = 0; l < 3; ++l) {
    dim3 g(HID / 64, NNODES / 32);
    if (l == 0)
      wmma_gemm_bf16<INCH, HID><<<g, 128, 0, stream>>>(xb, W0b, msg, nullptr, nullptr, 0);
    else
      wmma_gemm_bf16<HID, HID><<<g, 128, 0, stream>>>(xb, (l == 1) ? W1b : W2b, msg,
                                                      nullptr, nullptr, 0);

    gat_logits<<<nb(NNODES * HEADS, TB), TB, 0, stream>>>(msg, As[l], Ad[l], es, ed);
    fill_f32<<<nb(NNODES * HEADS, TB), TB, 0, stream>>>(mbuf, -1e30f, NNODES * HEADS);
    fill_f32<<<nb(NNODES * HEADS, TB), TB, 0, stream>>>(den, 0.f, NNODES * HEADS);
    fill_f32<<<nb((long long)NNODES * HID, TB), TB, 0, stream>>>(agg, 0.f, NNODES * HID);

    edge_max<<<nb((long long)ETOT * HEADS, TB), TB, 0, stream>>>(ei, es, ed, mbuf);
    edge_den<<<nb((long long)ETOT * HEADS, TB), TB, 0, stream>>>(ei, es, ed, mbuf, den);
    edge_agg<<<nb((long long)ETOT * HID, TB), TB, 0, stream>>>(ei, es, ed, mbuf, den, msg, agg);

    bias_act_cvt<<<nb((long long)NNODES * HID, TB), TB, 0, stream>>>(agg, Bi[l], xb, (l < 2) ? 1 : 0);
  }

  // Edge predictor MLP over (u,v) pairs; agg now holds final node embeddings (f32).
  gather_pairs<<<nb((long long)BATCH * 2 * HID, TB), TB, 0, stream>>>(agg, u, v, zb);
  {
    dim3 g(HID / 64, BATCH / 32);
    wmma_gemm_bf16<2 * HID, HID><<<g, 128, 0, stream>>>(zb, ew0b, z1f, z1b, eb0, 1);
  }
  {
    dim3 g((HID / 2) / 64, BATCH / 32);
    wmma_gemm_bf16<HID, HID / 2><<<g, 128, 0, stream>>>(z1b, ew1b, z2f, nullptr, eb1, 1);
  }
  edge_head<<<nb(BATCH, TB), TB, 0, stream>>>(z2f, ew2, eb2, u, v, out);
}
